// Model51_36215164240814
// MI455X (gfx1250) — compile-verified, matrix operand order checked by reading, not run on hardware
//
#include <hip/hip_runtime.h>
#include <hip/hip_bf16.h>
#include <cstdint>
#include <cstddef>

// ---------------- types for CDNA5 WMMA ----------------
typedef __bf16 bf16;
typedef __attribute__((ext_vector_type(16))) __bf16 v16bf;
typedef __attribute__((ext_vector_type(8)))  float  v8f;

#define WMMA_BF16(A, B, C) \
  __builtin_amdgcn_wmma_f32_16x16x32_bf16(false, (A), false, (B), (short)0, (C), false, false)

union Frag { v16bf v; uint4 q[2]; };

// ---------------- problem constants ----------------
#define BTc    8
#define Dc     256
#define NHc    8
#define HDc    32
#define LQc    5440
#define DFFNc  2048
#define MROWS  (BTc * LQc)          // 43520 = 340*128
#define M2ROWS (BTc * 128 * 128)    // 131072 = 1024*128
#define LOUT   21824                // 16384+4096+1024+256+64

__device__ __forceinline__ void qlevel(int q, int& l, int& p, int& s) {
  if (q < 4096)      { l = 0; p = q;        s = 64; }
  else if (q < 5120) { l = 1; p = q - 4096; s = 32; }
  else if (q < 5376) { l = 2; p = q - 5120; s = 16; }
  else               { l = 3; p = q - 5376; s = 8;  }
}

__device__ __forceinline__ bf16 tobf(float x) { return (bf16)x; }

// A fragment (16x32 bf16) from row-major [rows, ldk]; lane<16: K {0..7,16..23}, lane>=16: K {8..15,24..31}
__device__ __forceinline__ void frag_load(Frag& f, const bf16* __restrict__ base,
                                          int row0, int ldk, int k0, int lane) {
  const bf16* p = base + (size_t)(row0 + (lane & 15)) * ldk + (k0 + ((lane >> 4) << 3));
  f.q[0] = ((const uint4*)p)[0];
  f.q[1] = ((const uint4*)p)[2];   // +16 bf16 = +32 bytes
}

// ---------------- generic WMMA GEMM: C = A[M,K](bf16) * Wt[N,K](bf16)^T + bias ----------------
template <int RELU, int OUTBF16>
__global__ __launch_bounds__(256) void k_gemm(const bf16* __restrict__ A,
                                              const bf16* __restrict__ Wt,
                                              const float* __restrict__ bias,
                                              void* __restrict__ C,
                                              int M, int N, int K) {
  const int lane = threadIdx.x & 31;
  const int wv   = threadIdx.x >> 5;
  const int m0   = blockIdx.x * 128 + (wv & 3) * 32;
  const int n0   = blockIdx.y * 128 + (wv >> 2) * 64;
  v8f acc[2][4] = {};
  for (int k0 = 0; k0 < K; k0 += 32) {
    Frag a0, a1, b0, b1, b2, b3;
    frag_load(a0, A,  m0,      K, k0, lane);
    frag_load(a1, A,  m0 + 16, K, k0, lane);
    frag_load(b0, Wt, n0,      K, k0, lane);
    frag_load(b1, Wt, n0 + 16, K, k0, lane);
    frag_load(b2, Wt, n0 + 32, K, k0, lane);
    frag_load(b3, Wt, n0 + 48, K, k0, lane);
    acc[0][0] = WMMA_BF16(a0.v, b0.v, acc[0][0]);
    acc[0][1] = WMMA_BF16(a0.v, b1.v, acc[0][1]);
    acc[0][2] = WMMA_BF16(a0.v, b2.v, acc[0][2]);
    acc[0][3] = WMMA_BF16(a0.v, b3.v, acc[0][3]);
    acc[1][0] = WMMA_BF16(a1.v, b0.v, acc[1][0]);
    acc[1][1] = WMMA_BF16(a1.v, b1.v, acc[1][1]);
    acc[1][2] = WMMA_BF16(a1.v, b2.v, acc[1][2]);
    acc[1][3] = WMMA_BF16(a1.v, b3.v, acc[1][3]);
  }
  const int cn = lane & 15;
  const int rs = (lane >> 4) << 3;
  for (int i = 0; i < 2; ++i)
    for (int j = 0; j < 4; ++j) {
      int nn = n0 + j * 16 + cn;
      float bv = bias[nn];
#pragma unroll
      for (int r = 0; r < 8; ++r) {
        float x = acc[i][j][r] + bv;
        if (RELU) x = fmaxf(x, 0.f);
        size_t o = (size_t)(m0 + i * 16 + rs + r) * N + nn;
        if (OUTBF16) ((bf16*)C)[o] = tobf(x);
        else         ((float*)C)[o] = x;
      }
    }
}

// ---------------- shifted-A GEMM (one 3x3 conv tap), zero-padded border, optional accumulate ----------------
__global__ __launch_bounds__(256) void k_gemm_conv(const bf16* __restrict__ A,   // [B,H,W,C] bf16
                                                   const bf16* __restrict__ Wt,  // [N,K] bf16
                                                   const float* __restrict__ bias,
                                                   float* __restrict__ C,
                                                   int N, int K, int Hc, int Wc,
                                                   int dy, int dx, int accum) {
  const int lane = threadIdx.x & 31;
  const int wv   = threadIdx.x >> 5;
  const int m0   = blockIdx.x * 128 + (wv & 3) * 32;
  const int n0   = blockIdx.y * 128 + (wv >> 2) * 64;
  int rows[2]; bool ok[2];
#pragma unroll
  for (int i = 0; i < 2; ++i) {
    int m = m0 + i * 16 + (lane & 15);
    int b = m / (Hc * Wc);
    int rem = m - b * Hc * Wc;
    int y = rem / Wc + dy;
    int x = rem % Wc + dx;
    ok[i] = (y >= 0) && (y < Hc) && (x >= 0) && (x < Wc);
    rows[i] = (b * Hc + y) * Wc + x;
  }
  v8f acc[2][4] = {};
  for (int k0 = 0; k0 < K; k0 += 32) {
    Frag a0, a1, b0, b1, b2, b3;
    int kb = k0 + ((lane >> 4) << 3);
    if (ok[0]) { const bf16* p = A + (size_t)rows[0] * K + kb; a0.q[0] = ((const uint4*)p)[0]; a0.q[1] = ((const uint4*)p)[2]; }
    else       { a0.q[0] = make_uint4(0,0,0,0); a0.q[1] = make_uint4(0,0,0,0); }
    if (ok[1]) { const bf16* p = A + (size_t)rows[1] * K + kb; a1.q[0] = ((const uint4*)p)[0]; a1.q[1] = ((const uint4*)p)[2]; }
    else       { a1.q[0] = make_uint4(0,0,0,0); a1.q[1] = make_uint4(0,0,0,0); }
    frag_load(b0, Wt, n0,      K, k0, lane);
    frag_load(b1, Wt, n0 + 16, K, k0, lane);
    frag_load(b2, Wt, n0 + 32, K, k0, lane);
    frag_load(b3, Wt, n0 + 48, K, k0, lane);
    acc[0][0] = WMMA_BF16(a0.v, b0.v, acc[0][0]);
    acc[0][1] = WMMA_BF16(a0.v, b1.v, acc[0][1]);
    acc[0][2] = WMMA_BF16(a0.v, b2.v, acc[0][2]);
    acc[0][3] = WMMA_BF16(a0.v, b3.v, acc[0][3]);
    acc[1][0] = WMMA_BF16(a1.v, b0.v, acc[1][0]);
    acc[1][1] = WMMA_BF16(a1.v, b1.v, acc[1][1]);
    acc[1][2] = WMMA_BF16(a1.v, b2.v, acc[1][2]);
    acc[1][3] = WMMA_BF16(a1.v, b3.v, acc[1][3]);
  }
  const int cn = lane & 15;
  const int rs = (lane >> 4) << 3;
  for (int i = 0; i < 2; ++i)
    for (int j = 0; j < 4; ++j) {
      int nn = n0 + j * 16 + cn;
#pragma unroll
      for (int r = 0; r < 8; ++r) {
        size_t o = (size_t)(m0 + i * 16 + rs + r) * N + nn;
        float x = acc[i][j][r];
        if (accum) C[o] += x;
        else       C[o] = x + bias[nn];
      }
    }
}

// ---------------- pack src/pos (NCHW maps -> [M,256] channels-last) ----------------
__global__ void k_pack(const float* __restrict__ m1, const float* __restrict__ m2,
                       const float* __restrict__ m3, const float* __restrict__ m4,
                       const float* __restrict__ p1, const float* __restrict__ p2,
                       const float* __restrict__ p3, const float* __restrict__ p4,
                       const float* __restrict__ le,
                       float* __restrict__ src, float* __restrict__ pos) {
  int i = blockIdx.x * 256 + threadIdx.x;
  if (i >= MROWS * Dc) return;
  int c = i & 255, mq = i >> 8;
  int b = mq / LQc, q = mq - b * LQc;
  int l, p, s; qlevel(q, l, p, s);
  int hw = s * s;
  const float* ms[4] = {m1, m2, m3, m4};
  const float* ps[4] = {p1, p2, p3, p4};
  size_t si = ((size_t)(b * Dc + c)) * hw + p;
  src[i] = ms[l][si];
  pos[i] = ps[l][si] + le[l * Dc + c];
}

// ---------------- elementwise convert / add+convert ----------------
__global__ void k_cvt(const float* __restrict__ in, bf16* __restrict__ out, int n) {
  int i = blockIdx.x * 256 + threadIdx.x;
  if (i < n) out[i] = tobf(in[i]);
}
__global__ void k_addcvt(const float* __restrict__ a, const float* __restrict__ b,
                         bf16* __restrict__ out, int n) {
  int i = blockIdx.x * 256 + threadIdx.x;
  if (i < n) out[i] = tobf(a[i] + b[i]);
}

// ---------------- weight transpose+convert: in[K,N] f32 -> out[N,K] bf16 ----------------
__global__ void k_wtT(const float* __restrict__ in, bf16* __restrict__ out, int K, int N) {
  int i = blockIdx.x * 256 + threadIdx.x;
  if (i >= K * N) return;
  int k = i / N, n = i - k * N;
  out[(size_t)n * K + k] = tobf(in[i]);
}
// conv 3x3 weights (O,I,3,3) -> 9 tap matrices [N=O, K=I] bf16
__global__ void k_wcv(const float* __restrict__ in, bf16* __restrict__ out) {
  int i = blockIdx.x * 256 + threadIdx.x;
  if (i >= 9 * Dc * Dc) return;
  int t = i / (Dc * Dc);
  int oi = i - t * Dc * Dc;                   // o*256 + ich
  out[(size_t)t * Dc * Dc + oi] = tobf(in[(size_t)oi * 9 + t]);
}

// ---------------- softmax over groups of 16 ----------------
__global__ void k_softmax16(float* __restrict__ aw, int nrows) {
  int i = blockIdx.x * 256 + threadIdx.x;
  if (i >= nrows) return;
  float* p = aw + (size_t)i * 16;
  float mx = p[0];
  for (int j = 1; j < 16; ++j) mx = fmaxf(mx, p[j]);
  float e[16], s = 0.f;
  for (int j = 0; j < 16; ++j) { e[j] = __expf(p[j] - mx); s += e[j]; }
  float inv = 1.f / s;
  for (int j = 0; j < 16; ++j) p[j] = e[j] * inv;
}

// ---------------- deformable attention sampling: wave per (b,q,head), lane = head dim ----------------
__global__ __launch_bounds__(256) void k_deform(const float* __restrict__ vbuf,
                                                const float* __restrict__ off,
                                                const float* __restrict__ aw,
                                                float* __restrict__ acc) {
  const int gw = (blockIdx.x << 3) + (threadIdx.x >> 5);  // (b*LQ+q)*NH + h
  const int lane = threadIdx.x & 31;
  const int hh = gw & 7;
  const int mq = gw >> 3;
  const int b = mq / LQc;
  const int q = mq - b * LQc;
  int lq, pq, sq; qlevel(q, lq, pq, sq);
  const float gx = ((pq % sq) + 0.5f) / sq;
  const float gy = ((pq / sq) + 0.5f) / sq;
  const int c = hh * HDc + lane;
  const int st[4] = {0, 4096, 5120, 5376};
  const int sz[4] = {64, 32, 16, 8};
  float a = 0.f;
#pragma unroll
  for (int l2 = 0; l2 < 4; ++l2) {
    const int s = sz[l2], base = st[l2];
    const float fs = (float)s;
#pragma unroll
    for (int p = 0; p < 4; ++p) {
      int oi = (hh * 4 + l2) * 4 + p;
      float ox = off[(size_t)mq * 256 + oi * 2 + 0];
      float oy = off[(size_t)mq * 256 + oi * 2 + 1];
      float X = (gx + ox / fs) * fs - 0.5f;
      float Y = (gy + oy / fs) * fs - 0.5f;
      int x0 = (int)floorf(X), y0 = (int)floorf(Y);
      float wx = X - x0, wy = Y - y0;
      float smp = 0.f;
#pragma unroll
      for (int cy = 0; cy < 2; ++cy)
#pragma unroll
        for (int cx = 0; cx < 2; ++cx) {
          int yi = y0 + cy, xi = x0 + cx;
          if (yi >= 0 && yi < s && xi >= 0 && xi < s) {
            float w = (cy ? wy : 1.f - wy) * (cx ? wx : 1.f - wx);
            smp += w * vbuf[((size_t)(b * LQc + base + yi * s + xi)) * Dc + c];
          }
        }
      a += aw[(size_t)mq * 128 + oi] * smp;
    }
  }
  acc[(size_t)mq * Dc + c] = a;
}

// ---------------- residual + LayerNorm (block per row, D=256) ----------------
__global__ __launch_bounds__(256) void k_addln(float* __restrict__ src, const float* __restrict__ tmp,
                                               const float* __restrict__ g, const float* __restrict__ bta) {
  __shared__ float sh[256];
  const int row = blockIdx.x, d = threadIdx.x;
  const size_t idx = (size_t)row * Dc + d;
  const float v = src[idx] + tmp[idx];
  sh[d] = v; __syncthreads();
  for (int s = 128; s > 0; s >>= 1) { if (d < s) sh[d] += sh[d + s]; __syncthreads(); }
  const float mean = sh[0] * (1.f / Dc); __syncthreads();
  const float dv = v - mean;
  sh[d] = dv * dv; __syncthreads();
  for (int s = 128; s > 0; s >>= 1) { if (d < s) sh[d] += sh[d + s]; __syncthreads(); }
  const float var = sh[0] * (1.f / Dc);
  src[idx] = dv * rsqrtf(var + 1e-5f) * g[d] + bta[d];
}

// ---------------- ms0 NCHW -> channels-last bf16 ----------------
__global__ void k_pack_ms0(const float* __restrict__ ms0, bf16* __restrict__ out) {
  int i = blockIdx.x * 256 + threadIdx.x;
  if (i >= M2ROWS * Dc) return;
  int c = i & 255, bp = i >> 8;
  int b = bp >> 14, p = bp & 16383;
  out[i] = tobf(ms0[((size_t)(b * Dc + c)) * 16384 + p]);
}

// ---------------- bilinear 2x upsample of level-0 src region, add into large ----------------
__global__ void k_resize_add(float* __restrict__ large, const float* __restrict__ src) {
  int i = blockIdx.x * 256 + threadIdx.x;
  if (i >= M2ROWS * Dc) return;
  int c = i & 255, bp = i >> 8;
  int b = bp >> 14, p = bp & 16383;
  int y = p >> 7, x = p & 127;
  float sy = y * 0.5f - 0.25f, sx = x * 0.5f - 0.25f;
  int y0 = (int)floorf(sy), x0 = (int)floorf(sx);
  float wy = sy - y0, wx = sx - x0;
  int y0c = min(max(y0, 0), 63), y1c = min(max(y0 + 1, 0), 63);
  int x0c = min(max(x0, 0), 63), x1c = min(max(x0 + 1, 0), 63);
#define SRC0(yy, xx) src[((size_t)(b * LQc + (yy) * 64 + (xx))) * Dc + c]
  float v = (1.f - wy) * ((1.f - wx) * SRC0(y0c, x0c) + wx * SRC0(y0c, x1c)) +
            wy * ((1.f - wx) * SRC0(y1c, x0c) + wx * SRC0(y1c, x1c));
#undef SRC0
  large[i] += v;
}

// ---------------- GroupNorm (block per (b,group)) + affine + channel-major write into d_out ----------------
__global__ __launch_bounds__(256) void k_gn(const float* __restrict__ x, const float* __restrict__ g,
                                            const float* __restrict__ bta, float* __restrict__ out) {
  __shared__ float s1[256], s2[256];
  const int b = blockIdx.x >> 5, gr = blockIdx.x & 31;
  const int tid = threadIdx.x;
  float s = 0.f, ss = 0.f;
  for (int i = tid; i < 16384 * 8; i += 256) {
    int p = i >> 3, c0 = i & 7;
    float v = x[((size_t)(b * 16384 + p)) * Dc + gr * 8 + c0];
    s += v; ss += v * v;
  }
  s1[tid] = s; s2[tid] = ss; __syncthreads();
  for (int t = 128; t > 0; t >>= 1) { if (tid < t) { s1[tid] += s1[tid + t]; s2[tid] += s2[tid + t]; } __syncthreads(); }
  const float mean = s1[0] * (1.f / 131072.f);
  const float var  = s2[0] * (1.f / 131072.f) - mean * mean;
  const float rstd = rsqrtf(var + 1e-5f);
  for (int i = tid; i < 16384 * 8; i += 256) {
    int p = i >> 3, c = gr * 8 + (i & 7);
    float v = x[((size_t)(b * 16384 + p)) * Dc + c];
    out[((size_t)(b * Dc + c)) * LOUT + p] = (v - mean) * rstd * g[c] + bta[c];
  }
}

// ---------------- write encoder maps into d_out channel-major ----------------
__global__ void k_outs(const float* __restrict__ src, float* __restrict__ out) {
  int i = blockIdx.x * 256 + threadIdx.x;
  if (i >= MROWS * Dc) return;
  int c = i & 255, mq = i >> 8;
  int b = mq / LQc, q = mq - b * LQc;
  out[((size_t)(b * Dc + c)) * LOUT + 16384 + q] = src[i];
}

// ==================== host launcher ====================
extern "C" void kernel_launch(void* const* d_in, const int* in_sizes, int n_in,
                              void* d_out, int out_size, void* d_ws, size_t ws_size,
                              hipStream_t stream) {
  (void)in_sizes; (void)n_in; (void)out_size; (void)ws_size;
  const float* ms0 = (const float*)d_in[0];
  const float* ms1 = (const float*)d_in[1];
  const float* ms2 = (const float*)d_in[2];
  const float* ms3 = (const float*)d_in[3];
  const float* ms4 = (const float*)d_in[4];
  const float* pos1 = (const float*)d_in[5];
  const float* pos2 = (const float*)d_in[6];
  const float* pos3 = (const float*)d_in[7];
  const float* pos4 = (const float*)d_in[8];
  const float* level_embed = (const float*)d_in[9];
  const float* off_w = (const float*)d_in[10];
  const float* off_b = (const float*)d_in[11];
  const float* aw_w = (const float*)d_in[12];
  const float* aw_b = (const float*)d_in[13];
  const float* v_w = (const float*)d_in[14];
  const float* v_b = (const float*)d_in[15];
  const float* o_w = (const float*)d_in[16];
  const float* o_b = (const float*)d_in[17];
  const float* ln1_g = (const float*)d_in[18];
  const float* ln1_b = (const float*)d_in[19];
  const float* w1 = (const float*)d_in[20];
  const float* b1 = (const float*)d_in[21];
  const float* w2 = (const float*)d_in[22];
  const float* b2 = (const float*)d_in[23];
  const float* ln2_g = (const float*)d_in[24];
  const float* ln2_b = (const float*)d_in[25];
  const float* fpn_ad_w = (const float*)d_in[26];
  const float* fpn_ad_b = (const float*)d_in[27];
  const float* fpn_cv_w = (const float*)d_in[28];
  const float* fpn_cv_b = (const float*)d_in[29];
  const float* gn_g = (const float*)d_in[30];
  const float* gn_b = (const float*)d_in[31];
  float* outp = (float*)d_out;

  // -------- workspace bump allocator (deterministic) --------
  char* wp = (char*)d_ws;
  auto bump = [&](size_t bytes) -> char* {
    char* r = wp; wp += (bytes + 255) & ~((size_t)255); return r;
  };
  const size_t MD = (size_t)MROWS * Dc;
  float* src   = (float*)bump(MD * 4);
  float* pos   = (float*)bump(MD * 4);
  bf16*  srcbf = (bf16*)bump(MD * 2);
  bf16*  qbf   = (bf16*)bump(MD * 2);           // aliased as accbf
  float* offb  = (float*)bump(MD * 4);          // aliased as tmp
  float* awb   = (float*)bump((size_t)MROWS * 128 * 4);
  float* vbuf  = (float*)bump(MD * 4);
  float* accf  = (float*)bump(MD * 4);
  size_t bigsz = (size_t)MROWS * DFFNc * 2;     // 178MB >= convout 134MB
  char*  big   = bump(bigsz);
  bf16*  hbf   = (bf16*)big;
  float* convout = (float*)big;
  bf16*  ms0bf = (bf16*)bump((size_t)M2ROWS * Dc * 2);  // aliased as largebf
  float* large = (float*)bump((size_t)M2ROWS * Dc * 4);
  bf16*  wt_off = (bf16*)bump(65536 * 2);
  bf16*  wt_aw  = (bf16*)bump(32768 * 2);
  bf16*  wt_v   = (bf16*)bump(65536 * 2);
  bf16*  wt_o   = (bf16*)bump(65536 * 2);
  bf16*  wt_w1  = (bf16*)bump((size_t)524288 * 2);
  bf16*  wt_w2  = (bf16*)bump((size_t)524288 * 2);
  bf16*  wt_ad  = (bf16*)bump(65536 * 2);
  bf16*  wt_cv  = (bf16*)bump((size_t)9 * 65536 * 2);
  bf16*  accbf  = qbf;
  float* tmp    = offb;
  bf16*  largebf = ms0bf;

  auto gblk = [](size_t n) { return dim3((unsigned)((n + 255) / 256)); };

  // -------- pack src / pos --------
  k_pack<<<gblk(MD), 256, 0, stream>>>(ms1, ms2, ms3, ms4, pos1, pos2, pos3, pos4,
                                       level_embed, src, pos);

  // -------- encoder layers --------
  for (int i = 0; i < 2; ++i) {
    k_wtT<<<gblk(65536), 256, 0, stream>>>(off_w + (size_t)i * 65536, wt_off, 256, 256);
    k_wtT<<<gblk(32768), 256, 0, stream>>>(aw_w + (size_t)i * 32768, wt_aw, 256, 128);
    k_wtT<<<gblk(65536), 256, 0, stream>>>(v_w + (size_t)i * 65536, wt_v, 256, 256);
    k_wtT<<<gblk(65536), 256, 0, stream>>>(o_w + (size_t)i * 65536, wt_o, 256, 256);
    k_wtT<<<gblk(524288), 256, 0, stream>>>(w1 + (size_t)i * 524288, wt_w1, 256, 2048);
    k_wtT<<<gblk(524288), 256, 0, stream>>>(w2 + (size_t)i * 524288, wt_w2, 2048, 256);

    k_addcvt<<<gblk(MD), 256, 0, stream>>>(src, pos, qbf, (int)MD);
    k_cvt<<<gblk(MD), 256, 0, stream>>>(src, srcbf, (int)MD);

    k_gemm<0, 0><<<dim3(MROWS / 128, 2), 256, 0, stream>>>(qbf, wt_off, off_b + (size_t)i * 256, offb, MROWS, 256, 256);
    k_gemm<0, 0><<<dim3(MROWS / 128, 1), 256, 0, stream>>>(qbf, wt_aw, aw_b + (size_t)i * 128, awb, MROWS, 128, 256);
    k_softmax16<<<gblk((size_t)MROWS * NHc), 256, 0, stream>>>(awb, MROWS * NHc);
    k_gemm<0, 0><<<dim3(MROWS / 128, 2), 256, 0, stream>>>(srcbf, wt_v, v_b + (size_t)i * 256, vbuf, MROWS, 256, 256);

    k_deform<<<dim3(MROWS * NHc / 8), 256, 0, stream>>>(vbuf, offb, awb, accf);
    k_cvt<<<gblk(MD), 256, 0, stream>>>(accf, accbf, (int)MD);
    k_gemm<0, 0><<<dim3(MROWS / 128, 2), 256, 0, stream>>>(accbf, wt_o, o_b + (size_t)i * 256, tmp, MROWS, 256, 256);
    k_addln<<<dim3(MROWS), 256, 0, stream>>>(src, tmp, ln1_g + (size_t)i * 256, ln1_b + (size_t)i * 256);

    k_cvt<<<gblk(MD), 256, 0, stream>>>(src, srcbf, (int)MD);
    k_gemm<1, 1><<<dim3(MROWS / 128, 16), 256, 0, stream>>>(srcbf, wt_w1, b1 + (size_t)i * 2048, hbf, MROWS, 2048, 256);
    k_gemm<0, 0><<<dim3(MROWS / 128, 2), 256, 0, stream>>>(hbf, wt_w2, b2 + (size_t)i * 256, tmp, MROWS, 256, 2048);
    k_addln<<<dim3(MROWS), 256, 0, stream>>>(src, tmp, ln2_g + (size_t)i * 256, ln2_b + (size_t)i * 256);
  }

  // -------- FPN head --------
  k_cvt<<<gblk(65536), 256, 0, stream>>>(fpn_ad_w, wt_ad, 65536);  // (O,I,1,1) is already [N,K]
  k_wcv<<<gblk((size_t)9 * 65536), 256, 0, stream>>>(fpn_cv_w, wt_cv);
  k_pack_ms0<<<gblk((size_t)M2ROWS * Dc), 256, 0, stream>>>(ms0, ms0bf);
  k_gemm<0, 0><<<dim3(M2ROWS / 128, 2), 256, 0, stream>>>(ms0bf, wt_ad, fpn_ad_b, large, M2ROWS, 256, 256);
  k_resize_add<<<gblk((size_t)M2ROWS * Dc), 256, 0, stream>>>(large, src);
  k_cvt<<<gblk((size_t)M2ROWS * Dc), 256, 0, stream>>>(large, largebf, M2ROWS * Dc);
  for (int t = 0; t < 9; ++t) {
    int dy = t / 3 - 1, dx = t % 3 - 1;
    k_gemm_conv<<<dim3(M2ROWS / 128, 2), 256, 0, stream>>>(largebf, wt_cv + (size_t)t * 65536, fpn_cv_b,
                                                           convout, 256, 256, 128, 128, dy, dx, t != 0);
  }
  k_gn<<<dim3(BTc * 32), 256, 0, stream>>>(convout, gn_g, gn_b, outp);
  k_outs<<<gblk(MD), 256, 0, stream>>>(src, outp);
}
